// BBGRUDecoder_32839319945255
// MI455X (gfx1250) — matrix operand order repaired and checked by reading, not compiled, not run on hardware
//
#include <hip/hip_runtime.h>
#include <hip/hip_bf16.h>
#include <math.h>
#include <stdint.h>

// ---------------------------------------------------------------------------
// CDNA5 (gfx1250) wave32 WMMA types
// ---------------------------------------------------------------------------
typedef __attribute__((ext_vector_type(16))) __bf16 v16bf;
typedef __attribute__((ext_vector_type(8)))  __bf16 v8bf;
typedef __attribute__((ext_vector_type(8)))  float  v8f;

#define WMMA_BF16(A, B, C) \
    __builtin_amdgcn_wmma_f32_16x16x32_bf16(false, (A), false, (B), (short)0, (C), false, false)

// Problem constants
#define F1      128
#define F2      256
#define HID     128
#define TROUNDS 10
#define KOBS    12
#define NT_GRU  24      // 384 / 16 column tiles in GRU gate matrix

// ---------------------------------------------------------------------------
// Fragment layouts (CDNA5 ISA 16-bit WMMA, wave32):
//  A (16x32 MxK): lane = m + 16*h ; elem e: k = (e<8) ? 8h+e : 16+8h+(e-8)
//                 -> per lane: two contiguous 8-element (16B) runs
//  B (32x16 KxN): lane = n + 16*h ; elem e: k = (e<8) ? 8h+e : 16+8h+(e-8)
//  C/D (16x16 f32): vgpr r, lane l -> m = r + 8*(l>>4), n = l&15
// ---------------------------------------------------------------------------

// A fragment from a bf16 row-major panel staged in LDS (two ds_load_b128)
__device__ __forceinline__ v16bf load_a_lds(const __bf16* sA, int ld, int k0) {
    const int lane = threadIdx.x & 31;
    const int m = lane & 15, h = lane >> 4;
    const __bf16* p = sA + m * ld + k0;
    v8bf lo = *(const v8bf*)(p + 8 * h);
    v8bf hi = *(const v8bf*)(p + 16 + 8 * h);
    return __builtin_shufflevector(lo, hi, 0, 1, 2, 3, 4, 5, 6, 7,
                                   8, 9, 10, 11, 12, 13, 14, 15);
}

// B fragment from pre-packed fragment-order weights: one 32B contiguous load
__device__ __forceinline__ v16bf load_b_pk(const __bf16* __restrict__ pk, int tile) {
    const int lane = threadIdx.x & 31;
    return *(const v16bf*)(pk + ((size_t)tile * 32 + lane) * 16);
}

// ---------------------------------------------------------------------------
// 0) Weight pre-packing: fp32 W -> bf16 fragments in WMMA B order.
//    order=0: W[k*ld + n]  (conv weights, (K,N) row-major)
//    order=1: W[n*ld + k]  (torch GRU weights, (N,K) row-major)
//    One wave per 32x16 tile; lane writes its 16 contiguous elements.
// ---------------------------------------------------------------------------
__global__ void pack_b_kernel(const float* __restrict__ W, int ld, int K, int Nn,
                              int order, __bf16* __restrict__ pk) {
    const int gid  = blockIdx.x * blockDim.x + threadIdx.x;
    const int tile = gid >> 5;
    const int lane = gid & 31;
    const int ktiles = K >> 5, ntiles = Nn >> 4;
    if (tile >= ktiles * ntiles) return;
    const int kt = tile / ntiles, nt = tile % ntiles;
    const int n = lane & 15, h = lane >> 4;
    const int ncol = nt * 16 + n;
    __bf16* dst = pk + ((size_t)tile * 32 + lane) * 16;
#pragma unroll
    for (int e = 0; e < 16; ++e) {
        int k = kt * 32 + ((e < 8) ? 8 * h + e : 16 + 8 * h + (e - 8));
        float v = order ? W[(size_t)ncol * ld + k] : W[(size_t)k * ld + ncol];
        dst[e] = (__bf16)v;
    }
}

// ---------------------------------------------------------------------------
// 1) Edge scatter for conv1 (F_IN = 5)
// ---------------------------------------------------------------------------
__global__ void scatter1_kernel(const float* __restrict__ x,
                                const int* __restrict__ ei,
                                const float* __restrict__ ea,
                                float* __restrict__ agg1, int E) {
    int e = blockIdx.x * blockDim.x + threadIdx.x;
    if (e >= E) return;
    int s = ei[e], d = ei[E + e];
    float w = ea[e];
#pragma unroll
    for (int k = 0; k < 5; ++k)
        atomicAdd(&agg1[(size_t)d * 5 + k], x[(size_t)s * 5 + k] * w);
}

// ---------------------------------------------------------------------------
// 2) conv1 (K=5, VALU): h1 = relu(agg1@c1_wrel + x@c1_wroot + b), stored bf16
// ---------------------------------------------------------------------------
__global__ void conv1_kernel(const float* __restrict__ x,
                             const float* __restrict__ agg1,
                             const float* __restrict__ wrel,
                             const float* __restrict__ wroot,
                             const float* __restrict__ bias,
                             __bf16* __restrict__ h1bf) {
    int n = blockIdx.x;
    int f = threadIdx.x;                       // 0..127
    __shared__ float sx[5], sa[5];
    if (f < 5) { sx[f] = x[(size_t)n * 5 + f]; sa[f] = agg1[(size_t)n * 5 + f]; }
    __syncthreads();
    float acc = bias[f];
#pragma unroll
    for (int k = 0; k < 5; ++k)
        acc = fmaf(sa[k], wrel[k * F1 + f], fmaf(sx[k], wroot[k * F1 + f], acc));
    h1bf[(size_t)n * F1 + f] = (__bf16)fmaxf(acc, 0.f);
}

// ---------------------------------------------------------------------------
// 3) Edge scatter for conv2 (F1 = 128): one block per edge
// ---------------------------------------------------------------------------
__global__ void scatter2_kernel(const __bf16* __restrict__ h1bf,
                                const int* __restrict__ ei,
                                const float* __restrict__ ea,
                                float* __restrict__ agg2, int E) {
    int e = blockIdx.x;
    int k = threadIdx.x;                       // 0..127
    int s = ei[e], d = ei[E + e];
    float w = ea[e];
    atomicAdd(&agg2[(size_t)d * F1 + k], (float)h1bf[(size_t)s * F1 + k] * w);
}

// ---------------------------------------------------------------------------
// 4) Per-graph node counts
// ---------------------------------------------------------------------------
__global__ void count_kernel(const int* __restrict__ bl, float* __restrict__ gcnt, int N) {
    int n = blockIdx.x * blockDim.x + threadIdx.x;
    if (n < N) atomicAdd(&gcnt[bl[n]], 1.0f);
}

// ---------------------------------------------------------------------------
// 5) conv2 GEMM (WMMA bf16) fused with relu + mean-pool accumulation.
//    A panels staged once per block into LDS (fp32 converted at stage time);
//    B tiles come from pre-packed fragment-order weights (b128 loads).
// ---------------------------------------------------------------------------
__global__ __launch_bounds__(256)
void conv2_pool_kernel(const float*  __restrict__ agg2,
                       const __bf16* __restrict__ h1bf,
                       const __bf16* __restrict__ pk_wrel,   // 4x16 tiles
                       const __bf16* __restrict__ pk_wroot,
                       const float*  __restrict__ bias,
                       const int*    __restrict__ bl,
                       float* __restrict__ gsum, int N) {
    __shared__ __bf16 sA1[16 * F1];   // agg2 tile (converted)
    __shared__ __bf16 sA2[16 * F1];   // h1 tile
    const int row0 = blockIdx.x * 16;

    {   // cooperative staging: thread t -> row t/16, 8-elem chunk (t%16)*8
        const int m = threadIdx.x >> 4;
        const int c = (threadIdx.x & 15) * 8;
        int row = row0 + m; if (row >= N) row = N - 1;
        const float*  ga = agg2 + (size_t)row * F1 + c;
        const __bf16* gh = h1bf + (size_t)row * F1 + c;
#pragma unroll
        for (int u = 0; u < 8; ++u) {
            sA1[m * F1 + c + u] = (__bf16)ga[u];
            sA2[m * F1 + c + u] = gh[u];
        }
    }
    __syncthreads();

    const int wave = threadIdx.x >> 5;
    const int lane = threadIdx.x & 31;
    const int nlo  = lane & 15, hh = lane >> 4;

#pragma unroll
    for (int sub = 0; sub < 2; ++sub) {
        const int nt = wave * 2 + sub;         // column tile 0..15
        const int c0 = nt * 16;
        v8f acc = {};
#pragma unroll
        for (int kt = 0; kt < 4; ++kt) {
            if (kt < 3) {
                __builtin_prefetch(pk_wrel  + ((size_t)((kt + 1) * 16 + nt) * 32) * 16, 0, 1);
                __builtin_prefetch(pk_wroot + ((size_t)((kt + 1) * 16 + nt) * 32) * 16, 0, 1);
            }
            v16bf a1 = load_a_lds(sA1, F1, kt * 32);
            acc = WMMA_BF16(a1, load_b_pk(pk_wrel, kt * 16 + nt), acc);
            v16bf a2 = load_a_lds(sA2, F1, kt * 32);
            acc = WMMA_BF16(a2, load_b_pk(pk_wroot, kt * 16 + nt), acc);
        }
        const float bv = bias[c0 + nlo];
#pragma unroll
        for (int r = 0; r < 8; ++r) {
            int node = row0 + r + 8 * hh;
            if (node < N) {
                float v = fmaxf(acc[r] + bv, 0.f);
                atomicAdd(&gsum[(size_t)bl[node] * F2 + c0 + nlo], v);
            }
        }
    }
}

// ---------------------------------------------------------------------------
// 6) bulk (bf16, [T][B][F2]) init + per-graph mean scatter + active mask
// ---------------------------------------------------------------------------
__global__ void bulk_init_kernel(const float* __restrict__ empty_emb,
                                 __bf16* __restrict__ bulk, int total) {
    int i = blockIdx.x * blockDim.x + threadIdx.x;
    if (i < total) bulk[i] = (__bf16)empty_emb[i & (F2 - 1)];
}

__global__ void emb_scatter_kernel(const float* __restrict__ gsum,
                                   const float* __restrict__ gcnt,
                                   const int* __restrict__ lm,
                                   __bf16* __restrict__ bulk, int Bsz) {
    int g = blockIdx.x;
    int f = threadIdx.x;                       // 0..255
    float c = fmaxf(gcnt[g], 1.0f);
    float v = gsum[(size_t)g * F2 + f] / c;
    int shot = lm[g * 2 + 0];
    int rnd  = lm[g * 2 + 1];
    bulk[((size_t)rnd * Bsz + shot) * F2 + f] = (__bf16)v;
}

__global__ void active_kernel(const int* __restrict__ lm, int* __restrict__ act, int G) {
    int g = blockIdx.x * blockDim.x + threadIdx.x;
    if (g < G) act[lm[g * 2]] = 1;
}

// ---------------------------------------------------------------------------
// 7) GRU step (one layer, one timestep), fused gates via WMMA.
//    X / Hprev panels staged in LDS (bf16 straight copies), packed weights.
//    Writes both fp32 state (for gating/decode) and bf16 mirror (next A).
// ---------------------------------------------------------------------------
__global__ __launch_bounds__(256)
void gru_step_kernel(const __bf16* __restrict__ X, int Kx,
                     const float*  __restrict__ Hprev,
                     const __bf16* __restrict__ Hprevbf,
                     const __bf16* __restrict__ pk_ih,   // (Kx/32) x 24 tiles
                     const __bf16* __restrict__ pk_hh,   // 4 x 24 tiles
                     const float* __restrict__ bih,
                     const float* __restrict__ bhh,
                     float*  __restrict__ Hnew,
                     __bf16* __restrict__ Hnewbf, int Bsz) {
    __shared__ __bf16 sX[16 * F2];     // up to Kx = 256
    __shared__ __bf16 sH[16 * HID];
    const int row0 = blockIdx.x * 16;  // Bsz is a multiple of 16

    {   // straight bf16 panel copies as dwords (rows are contiguous)
        const uint32_t* gx = (const uint32_t*)(X + (size_t)row0 * Kx);
        const uint32_t* gh = (const uint32_t*)(Hprevbf + (size_t)row0 * HID);
        uint32_t* dx = (uint32_t*)sX;
        uint32_t* dh = (uint32_t*)sH;
        for (int i = threadIdx.x; i < 16 * Kx / 2; i += 256) dx[i] = gx[i];
        for (int i = threadIdx.x; i < 16 * HID / 2; i += 256) dh[i] = gh[i];
    }
    __syncthreads();

    const int wave = threadIdx.x >> 5;
    const int j0   = wave * 16;
    const int lane = threadIdx.x & 31;
    const int nlo  = lane & 15, hh = lane >> 4;

    v8f acc_r = {}, acc_z = {}, acc_in = {}, acc_hn = {};

    const int kts = Kx >> 5;
    for (int kt = 0; kt < kts; ++kt) {          // input-side gates
        if (kt + 1 < kts)
            __builtin_prefetch(pk_ih + ((size_t)((kt + 1) * NT_GRU + wave) * 32) * 16, 0, 1);
        v16bf a = load_a_lds(sX, Kx, kt * 32);
        acc_r  = WMMA_BF16(a, load_b_pk(pk_ih, kt * NT_GRU + 0 * 8 + wave), acc_r);
        acc_z  = WMMA_BF16(a, load_b_pk(pk_ih, kt * NT_GRU + 1 * 8 + wave), acc_z);
        acc_in = WMMA_BF16(a, load_b_pk(pk_ih, kt * NT_GRU + 2 * 8 + wave), acc_in);
    }
#pragma unroll
    for (int kt = 0; kt < 4; ++kt) {            // hidden-side gates
        v16bf a = load_a_lds(sH, HID, kt * 32);
        acc_r  = WMMA_BF16(a, load_b_pk(pk_hh, kt * NT_GRU + 0 * 8 + wave), acc_r);
        acc_z  = WMMA_BF16(a, load_b_pk(pk_hh, kt * NT_GRU + 1 * 8 + wave), acc_z);
        acc_hn = WMMA_BF16(a, load_b_pk(pk_hh, kt * NT_GRU + 2 * 8 + wave), acc_hn);
    }

    const int j = j0 + nlo;
    const float br_ = bih[j] + bhh[j];
    const float bz_ = bih[HID + j] + bhh[HID + j];
    const float bin = bih[2 * HID + j];
    const float bhn = bhh[2 * HID + j];
#pragma unroll
    for (int r = 0; r < 8; ++r) {
        int b = row0 + r + 8 * hh;
        float rg = 1.0f / (1.0f + __expf(-(acc_r[r] + br_)));
        float zg = 1.0f / (1.0f + __expf(-(acc_z[r] + bz_)));
        float ng = tanhf(acc_in[r] + bin + rg * (acc_hn[r] + bhn));
        float hp = Hprev[(size_t)b * HID + j];
        float hn = (1.0f - zg) * ng + zg * hp;
        Hnew[(size_t)b * HID + j]   = hn;
        Hnewbf[(size_t)b * HID + j] = (__bf16)hn;
    }
}

// ---------------------------------------------------------------------------
// 8) Decoder: logits = h_last @ dec_w + dec_b, zeroed for inactive shots
// ---------------------------------------------------------------------------
__global__ void decoder_kernel(const float* __restrict__ hlast,
                               const float* __restrict__ dw,   // [128,12]
                               const float* __restrict__ db,   // [12]
                               const int* __restrict__ act,
                               float* __restrict__ out, int total) {
    int i = blockIdx.x * blockDim.x + threadIdx.x;
    if (i >= total) return;
    int b = i / KOBS, k = i % KOBS;
    float acc = db[k];
#pragma unroll 8
    for (int j = 0; j < HID; ++j)
        acc = fmaf(hlast[(size_t)b * HID + j], dw[j * KOBS + k], acc);
    out[i] = act[b] ? acc : 0.0f;
}

// ---------------------------------------------------------------------------
// Host-side launcher
// ---------------------------------------------------------------------------
extern "C" void kernel_launch(void* const* d_in, const int* in_sizes, int n_in,
                              void* d_out, int out_size, void* d_ws, size_t ws_size,
                              hipStream_t stream) {
    const float* x         = (const float*)d_in[0];
    const int*   ei        = (const int*)  d_in[1];
    const float* ea        = (const float*)d_in[2];
    const int*   bl        = (const int*)  d_in[3];
    const int*   lm        = (const int*)  d_in[4];
    const float* c1_wrel   = (const float*)d_in[6];
    const float* c1_wroot  = (const float*)d_in[7];
    const float* c1_b      = (const float*)d_in[8];
    const float* c2_wrel   = (const float*)d_in[9];
    const float* c2_wroot  = (const float*)d_in[10];
    const float* c2_b      = (const float*)d_in[11];
    const float* empty_emb = (const float*)d_in[12];
    const float* w_ih0     = (const float*)d_in[13];
    const float* w_hh0     = (const float*)d_in[14];
    const float* b_ih0     = (const float*)d_in[15];
    const float* b_hh0     = (const float*)d_in[16];
    const float* w_ih1     = (const float*)d_in[17];
    const float* w_hh1     = (const float*)d_in[18];
    const float* b_ih1     = (const float*)d_in[19];
    const float* b_hh1     = (const float*)d_in[20];
    const float* dec_w     = (const float*)d_in[21];
    const float* dec_b     = (const float*)d_in[22];

    const int N   = in_sizes[0] / 5;        // 100000 nodes
    const int E   = in_sizes[2];            // 300000 edges
    const int NG  = in_sizes[4] / 2;        // 8192 graphs
    const int Bsz = out_size / KOBS;        // 1024 shots

    // ---- workspace carve-out (256B-aligned chunks) ----
    char* base = (char*)d_ws;
    size_t off = 0;
    auto alloc = [&](size_t bytes) -> char* {
        char* p = base + off;
        off = (off + bytes + 255) & ~(size_t)255;
        return p;
    };

    // zero region A: scatter/pool accumulators
    float* agg1 = (float*)alloc((size_t)N * 5 * 4);
    float* agg2 = (float*)alloc((size_t)N * F1 * 4);
    float* gsum = (float*)alloc((size_t)NG * F2 * 4);
    float* gcnt = (float*)alloc((size_t)NG * 4);
    const size_t zeroA_bytes = off;

    __bf16* h1bf   = (__bf16*)alloc((size_t)N * F1 * 2);
    __bf16* bulkbf = (__bf16*)alloc((size_t)TROUNDS * Bsz * F2 * 2);

    __bf16* pk_wrel  = (__bf16*)alloc((size_t)4 * 16 * 512 * 2);      // (K/32)x(N/16)x512
    __bf16* pk_wroot = (__bf16*)alloc((size_t)4 * 16 * 512 * 2);
    __bf16* pk_ih0   = (__bf16*)alloc((size_t)8 * NT_GRU * 512 * 2);
    __bf16* pk_hh0   = (__bf16*)alloc((size_t)4 * NT_GRU * 512 * 2);
    __bf16* pk_ih1   = (__bf16*)alloc((size_t)4 * NT_GRU * 512 * 2);
    __bf16* pk_hh1   = (__bf16*)alloc((size_t)4 * NT_GRU * 512 * 2);

    // zero region B: GRU states (fp32 + bf16 mirrors) and active mask
    const size_t zeroB_off = off;
    float* s0[2]; float* s1[2]; __bf16* s0b[2]; __bf16* s1b[2];
    s0[0]  = (float*)alloc((size_t)Bsz * HID * 4);
    s0[1]  = (float*)alloc((size_t)Bsz * HID * 4);
    s1[0]  = (float*)alloc((size_t)Bsz * HID * 4);
    s1[1]  = (float*)alloc((size_t)Bsz * HID * 4);
    s0b[0] = (__bf16*)alloc((size_t)Bsz * HID * 2);
    s0b[1] = (__bf16*)alloc((size_t)Bsz * HID * 2);
    s1b[0] = (__bf16*)alloc((size_t)Bsz * HID * 2);
    s1b[1] = (__bf16*)alloc((size_t)Bsz * HID * 2);
    int* act = (int*)alloc((size_t)Bsz * 4);
    const size_t zeroB_bytes = off - zeroB_off;
    (void)ws_size; (void)n_in;

    hipMemsetAsync(base, 0, zeroA_bytes, stream);
    hipMemsetAsync(base + zeroB_off, 0, zeroB_bytes, stream);

    // ---- one-time weight packing (fp32 -> bf16 WMMA-B fragment order) ----
    auto packGrid = [](int K, int Nn) { return ((K / 32) * (Nn / 16) * 32 + 255) / 256; };
    pack_b_kernel<<<packGrid(F1, F2), 256, 0, stream>>>(c2_wrel,  F2,  F1, F2,  0, pk_wrel);
    pack_b_kernel<<<packGrid(F1, F2), 256, 0, stream>>>(c2_wroot, F2,  F1, F2,  0, pk_wroot);
    pack_b_kernel<<<packGrid(F2, 384), 256, 0, stream>>>(w_ih0, F2,  F2,  384, 1, pk_ih0);
    pack_b_kernel<<<packGrid(HID, 384), 256, 0, stream>>>(w_hh0, HID, HID, 384, 1, pk_hh0);
    pack_b_kernel<<<packGrid(HID, 384), 256, 0, stream>>>(w_ih1, HID, HID, 384, 1, pk_ih1);
    pack_b_kernel<<<packGrid(HID, 384), 256, 0, stream>>>(w_hh1, HID, HID, 384, 1, pk_hh1);

    // ---- GNN stage ----
    scatter1_kernel<<<(E + 255) / 256, 256, 0, stream>>>(x, ei, ea, agg1, E);
    conv1_kernel<<<N, F1, 0, stream>>>(x, agg1, c1_wrel, c1_wroot, c1_b, h1bf);
    scatter2_kernel<<<E, F1, 0, stream>>>(h1bf, ei, ea, agg2, E);
    count_kernel<<<(N + 255) / 256, 256, 0, stream>>>(bl, gcnt, N);
    conv2_pool_kernel<<<(N + 15) / 16, 256, 0, stream>>>(agg2, h1bf, pk_wrel, pk_wroot,
                                                         c2_b, bl, gsum, N);

    // ---- group into [T, B, F2] (bf16) ----
    const int bulkTotal = TROUNDS * Bsz * F2;
    bulk_init_kernel<<<(bulkTotal + 255) / 256, 256, 0, stream>>>(empty_emb, bulkbf, bulkTotal);
    emb_scatter_kernel<<<NG, F2, 0, stream>>>(gsum, gcnt, lm, bulkbf, Bsz);
    active_kernel<<<(NG + 255) / 256, 256, 0, stream>>>(lm, act, NG);

    // ---- two-layer GRU over T timesteps (double-buffered states) ----
    int cur = 0;
    const int gruGrid = Bsz / 16;
    for (int t = 0; t < TROUNDS; ++t) {
        const __bf16* xt = bulkbf + (size_t)t * Bsz * F2;
        gru_step_kernel<<<gruGrid, 256, 0, stream>>>(xt, F2, s0[cur], s0b[cur],
                                                     pk_ih0, pk_hh0, b_ih0, b_hh0,
                                                     s0[1 - cur], s0b[1 - cur], Bsz);
        gru_step_kernel<<<gruGrid, 256, 0, stream>>>(s0b[1 - cur], HID, s1[cur], s1b[cur],
                                                     pk_ih1, pk_hh1, b_ih1, b_hh1,
                                                     s1[1 - cur], s1b[1 - cur], Bsz);
        cur ^= 1;
    }
    const float* hlast = s1[cur];

    // ---- decode ----
    decoder_kernel<<<(Bsz * KOBS + 255) / 256, 256, 0, stream>>>(hlast, dec_w, dec_b, act,
                                                                 (float*)d_out, Bsz * KOBS);
}
